// GNN_56272661512637
// MI455X (gfx1250) — matrix operand order repaired and checked by reading, not compiled, not run on hardware
//
#include <hip/hip_runtime.h>
#include <hip/hip_bf16.h>

typedef __attribute__((ext_vector_type(16))) __bf16 v16bf;
typedef __attribute__((ext_vector_type(8)))  float  v8f;
typedef __attribute__((ext_vector_type(4)))  float  v4f;
typedef const __attribute__((address_space(1))) v4f* gc4;  // force global_load path

#define LATENT  256
#define NUM_OUT 128
#define Nn      65536
#define Ne      131072
#define Gg      128
#define NSTEPS  5

// ---------------------------------------------------------------- utilities
__global__ __launch_bounds__(256) void fill_zero(float* __restrict__ p, size_t n) {
  size_t i = (size_t)blockIdx.x * 256 + threadIdx.x;
  if (i < n) p[i] = 0.f;
}

// out[r,c] = sum_k feats[r,k]*W[k,c] + b[c]   (tiny K: 9 or 3)
__global__ __launch_bounds__(256) void embed_k(const float* __restrict__ f, int F,
    const float* __restrict__ W, const float* __restrict__ b,
    float* __restrict__ out, int M) {
  int gid = blockIdx.x * 256 + threadIdx.x;
  if (gid >= M * LATENT) return;
  int r = gid >> 8, c = gid & 255;
  float acc = b[c];
  for (int k = 0; k < F; ++k) acc += f[r * F + k] * W[k * LATENT + c];
  out[gid] = acc;
}

// Swizzle fp32 weights [Din,Nout] into per-lane-contiguous bf16 B fragments.
// B 32x16 bf16 layout used here (contiguous k, same as A staging): lanes 0-15 hold K=0..15, lanes 16-31 hold K=16..31;
// element e of the v16bf = K&15.
__global__ __launch_bounds__(256) void pack_b_k(const float* __restrict__ W,
    __bf16* __restrict__ Bp, int Din, int Nout) {
  int gid = blockIdx.x * 256 + threadIdx.x;
  if (gid >= Din * Nout) return;
  int K = gid / Nout, n = gid - K * Nout;
  int stage = K >> 5, k32 = K & 31;
  int lane = (n & 15) | (k32 & 16);
  int e = k32 & 15;
  int ntiles = Nout >> 4;
  Bp[((size_t)(stage * ntiles + (n >> 4)) * 32 + lane) * 16 + e] = (__bf16)W[gid];
}

// ------------------------------------------------- fused gather-concat GEMM
// H[M, CT*128] = concat(seg0..seg3)[M,Din] @ Wpacked.
// Block: 32 rows (2 row tiles) x NOUT cols, 8 waves; wave wv owns col tiles
// {wv, wv+8}[0..CT). Per-thread segment base addresses are computed once;
// the inner loop advances one running pointer (+64 floats) with a uniform
// boundary test. Double-buffered LDS A staging (1 barrier/chunk), 4*CT WMMAs
// per 64-wide chunk per wave over 2*CT independent accumulator chains.
template <int CT>
__global__ __launch_bounds__(256, 1) void gemm_fused(
    const float* __restrict__ p0, const int* __restrict__ i0, int w0,
    const float* __restrict__ p1, const int* __restrict__ i1, int w1,
    const float* __restrict__ p2, const int* __restrict__ i2, int w2,
    const float* __restrict__ p3, const int* __restrict__ i3, int w3,
    const __bf16* __restrict__ Bp, float* __restrict__ H, int Din) {
  constexpr int NT = CT * 8;          // total col tiles
  constexpr int NOUT = NT * 16;
  constexpr size_t FRAG = (size_t)NT * 512;  // bf16 elems per K-stage

  const int rowBase = blockIdx.x << 5;
  const int tid = threadIdx.x;
  const int lane = tid & 31;
  const int wv = tid >> 5;

  __shared__ int rmap[4][32];
  __shared__ __bf16 As[2][2][2][32][16] __attribute__((aligned(32))); // [buf][rt][sub]

  if (tid < 128) {
    int s = tid >> 5, r = tid & 31;
    const int* idx = (s == 0) ? i0 : (s == 1) ? i1 : (s == 2) ? i2 : i3;
    rmap[s][r] = idx ? idx[rowBase + r] : (rowBase + r);
  }
  __syncthreads();

  const int b1 = w0, b2 = w0 + w1, b3 = w0 + w1 + w2;

  // Per-thread A staging coordinates: thread t stages row srow, cols c0..c0+7.
  // Those 8 bf16 land contiguously in the A-fragment layout:
  //   lane = (srow&15) | (h<<4), h=(c0>>4)&1; e0 = c0&8 (0 or 8).
  const int srow = tid >> 3;          // 0..31
  const int c0 = (tid & 7) << 3;      // 0,8,...,56
  const int s_rt = srow >> 4;
  const int s_sub = (c0 >> 5) & 1;
  const int s_lane = (srow & 15) | (c0 & 16);
  const int s_e0 = c0 & 8;

  // Segment base addresses for this thread's staging row (computed once).
  const float* q0 = p0 + (size_t)rmap[0][srow] * w0 + c0;
  const float* q1 = w1 ? (p1 + (size_t)rmap[1][srow] * w1 + c0) : q0;
  const float* q2 = w2 ? (p2 + (size_t)rmap[2][srow] * w2 + c0) : q0;
  const float* q3 = w3 ? (p3 + (size_t)rmap[3][srow] * w3 + c0) : q0;

  v4f rva, rvb;
  const float* cur = q0;
  rva = *(gc4)cur;
  rvb = *(gc4)(cur + 4);

  auto commit = [&](int buf) {
    union { __bf16 h[8]; v4f f4; } pk;
    pk.h[0] = (__bf16)rva.x; pk.h[1] = (__bf16)rva.y;
    pk.h[2] = (__bf16)rva.z; pk.h[3] = (__bf16)rva.w;
    pk.h[4] = (__bf16)rvb.x; pk.h[5] = (__bf16)rvb.y;
    pk.h[6] = (__bf16)rvb.z; pk.h[7] = (__bf16)rvb.w;
    *(v4f*)(&As[buf][s_rt][s_sub][s_lane][s_e0]) = pk.f4;
  };

  v8f acc[2][CT];
#pragma unroll
  for (int rt = 0; rt < 2; ++rt)
#pragma unroll
    for (int ct = 0; ct < CT; ++ct) acc[rt][ct] = (v8f){};

  const __bf16* bptr = Bp + ((size_t)wv * 32 + lane) * 16;

  // Two K-sub phases per chunk: halves fragment register liveness.
  auto domma = [&](int buf) {
#pragma unroll
    for (int sub = 0; sub < 2; ++sub) {
      v16bf a0 = *(const v16bf*)(&As[buf][0][sub][lane][0]);
      v16bf a1 = *(const v16bf*)(&As[buf][1][sub][lane][0]);
      v16bf bb0 = *(const v16bf*)(bptr + (size_t)sub * FRAG);
      acc[0][0] = __builtin_amdgcn_wmma_f32_16x16x32_bf16(
          false, a0, false, bb0, (short)0, acc[0][0], false, false);
      acc[1][0] = __builtin_amdgcn_wmma_f32_16x16x32_bf16(
          false, a1, false, bb0, (short)0, acc[1][0], false, false);
      if constexpr (CT == 2) {
        v16bf bb1 = *(const v16bf*)(bptr + (8 * 512) + (size_t)sub * FRAG);
        acc[0][1] = __builtin_amdgcn_wmma_f32_16x16x32_bf16(
            false, a0, false, bb1, (short)0, acc[0][1], false, false);
        acc[1][1] = __builtin_amdgcn_wmma_f32_16x16x32_bf16(
            false, a1, false, bb1, (short)0, acc[1][1], false, false);
      }
    }
    bptr += 2 * FRAG;
  };

  // Software pipeline: regs for chunk k+1 load while WMMAs of chunk k run.
  int buf = 0;
  const int lastKb = Din - 64;
  for (int kb = 0; kb < lastKb; kb += 64) {
    commit(buf);
    __syncthreads();
    const int kn = kb + 64;              // advance running pointer (uniform test)
    cur = (kn == b1) ? q1 : (kn == b2) ? q2 : (kn == b3) ? q3 : (cur + 64);
    rva = *(gc4)cur;
    rvb = *(gc4)(cur + 4);
    domma(buf);
    buf ^= 1;
  }
  commit(buf);
  __syncthreads();
  domma(buf);

  // C/D layout: lane 0-15 => N=lane, VGPR r => M=r; lane 16-31 => M=8+r
  const int colb = lane & 15;
  const int rowoff = (lane >> 4) << 3;
#pragma unroll
  for (int ct = 0; ct < CT; ++ct) {
    float* hp = H + (size_t)(rowBase + rowoff) * NOUT + ((wv + ct * 8) << 4) + colb;
#pragma unroll
    for (int rt = 0; rt < 2; ++rt)
#pragma unroll
      for (int r = 0; r < 8; ++r)
        hp[(size_t)(rt * 16 + r) * NOUT] = acc[rt][ct][r];
  }
}

// ------------------------------------------------ bias + LayerNorm + ReLU
__global__ __launch_bounds__(256) void ln_relu_k(const float* __restrict__ H,
    const float* __restrict__ bias, const float* __restrict__ g,
    const float* __restrict__ bln, float* __restrict__ dst) {
  const int row = blockIdx.x, c = threadIdx.x;
  float x = H[(size_t)row * LATENT + c] + bias[c];
  __shared__ float red[9];
  float s = x;
#pragma unroll
  for (int o = 16; o; o >>= 1) s += __shfl_down(s, o, 32);
  if ((c & 31) == 0) red[c >> 5] = s;
  __syncthreads();
  if (c == 0) { float t = 0; for (int i = 0; i < 8; ++i) t += red[i]; red[8] = t * (1.f / LATENT); }
  __syncthreads();
  const float m = red[8];
  float d = x - m, s2 = d * d;
#pragma unroll
  for (int o = 16; o; o >>= 1) s2 += __shfl_down(s2, o, 32);
  __syncthreads();
  if ((c & 31) == 0) red[c >> 5] = s2;
  __syncthreads();
  if (c == 0) { float t = 0; for (int i = 0; i < 8; ++i) t += red[i]; red[8] = t * (1.f / LATENT); }
  __syncthreads();
  float inv = rsqrtf(red[8] + 1e-6f);
  float y = d * inv * g[c] + bln[c];
  dst[(size_t)row * LATENT + c] = fmaxf(y, 0.f);
}

// segment_sum(edges -> per-node) via f32 atomics
__global__ __launch_bounds__(256) void scatter_k(const float* __restrict__ edges,
    const int* __restrict__ snd, const int* __restrict__ rcv,
    float* __restrict__ sacc, float* __restrict__ racc) {
  size_t gid = (size_t)blockIdx.x * 256 + threadIdx.x;
  if (gid >= (size_t)Ne * LATENT) return;
  int e = (int)(gid >> 8), c = (int)(gid & 255);
  float v = edges[gid];
  atomicAdd(&sacc[(size_t)snd[e] * LATENT + c], v);
  atomicAdd(&racc[(size_t)rcv[e] * LATENT + c], v);
}

// per-graph sum over contiguous slices (graphs are equal contiguous blocks)
__global__ __launch_bounds__(256) void gsum_k(const float* __restrict__ src,
    float* __restrict__ dst, int per) {
  int gidx = blockIdx.x, c = threadIdx.x;
  const float* base = src + (size_t)gidx * per * LATENT;
  float acc = 0.f;
  for (int i = 0; i < per; ++i) acc += base[(size_t)i * LATENT + c];
  dst[(size_t)gidx * LATENT + c] = acc;
}

__global__ __launch_bounds__(256) void bias_out_k(const float* __restrict__ H,
    const float* __restrict__ b, float* __restrict__ out, int n, int Nout) {
  int gid = blockIdx.x * 256 + threadIdx.x;
  if (gid < n) out[gid] = H[gid] + b[gid % Nout];
}

// ---------------------------------------------------------------- launcher
extern "C" void kernel_launch(void* const* d_in, const int* in_sizes, int n_in,
                              void* d_out, int out_size, void* d_ws, size_t ws_size,
                              hipStream_t stream) {
  const float* node_feats = (const float*)d_in[0];
  const float* edge_feats = (const float*)d_in[1];
  const int* senders   = (const int*)d_in[2];
  const int* receivers = (const int*)d_in[3];
  const int* node_gidx = (const int*)d_in[4];
  const int* edge_gidx = (const int*)d_in[5];
  auto P = [&](int i) { return (const float*)d_in[i]; };

  char* ws = (char*)d_ws;
  size_t off = 0;
  auto alloc = [&](size_t bytes) -> char* {
    char* p = ws + off;
    off = (off + bytes + 255) & ~(size_t)255;
    return p;
  };
  float* nodes = (float*)alloc((size_t)Nn * LATENT * 4);
  float* edges = (float*)alloc((size_t)Ne * LATENT * 4);
  float* glob  = (float*)alloc((size_t)Gg * LATENT * 4);
  float* Hbuf  = (float*)alloc((size_t)Ne * LATENT * 4);
  float* sacc  = (float*)alloc((size_t)Nn * LATENT * 4);
  float* racc  = (float*)alloc((size_t)Nn * LATENT * 4);
  float* nagg  = (float*)alloc((size_t)Gg * LATENT * 4);
  float* eagg  = (float*)alloc((size_t)Gg * LATENT * 4);

  __bf16* BpE[NSTEPS]; __bf16* BpN[NSTEPS]; __bf16* BpG[NSTEPS]; __bf16* BpD;
  int DinE[NSTEPS], DinG[NSTEPS];
  for (int t = 0; t < NSTEPS; ++t) {
    int gdim = (t == 0) ? NUM_OUT : LATENT;
    DinE[t] = 3 * LATENT + gdim;
    DinG[t] = 2 * LATENT + gdim;
    BpE[t] = (__bf16*)alloc((size_t)DinE[t] * LATENT * 2);
    BpN[t] = (__bf16*)alloc((size_t)DinE[t] * LATENT * 2);
    BpG[t] = (__bf16*)alloc((size_t)DinG[t] * LATENT * 2);
  }
  BpD = (__bf16*)alloc((size_t)LATENT * NUM_OUT * 2);

  // globals start at zero ([G,128] packed inside a [G,256] buffer)
  fill_zero<<<(Gg * LATENT + 255) / 256, 256, 0, stream>>>(glob, (size_t)Gg * LATENT);

  // pack all weights into WMMA B-fragment layout (bf16)
  for (int t = 0; t < NSTEPS; ++t) {
    int base = 10 + t * 12;
    int ne = DinE[t] * LATENT, ng = DinG[t] * LATENT;
    pack_b_k<<<(ne + 255) / 256, 256, 0, stream>>>(P(base + 0), BpE[t], DinE[t], LATENT);
    pack_b_k<<<(ne + 255) / 256, 256, 0, stream>>>(P(base + 4), BpN[t], DinE[t], LATENT);
    pack_b_k<<<(ng + 255) / 256, 256, 0, stream>>>(P(base + 8), BpG[t], DinG[t], LATENT);
  }
  pack_b_k<<<((LATENT * NUM_OUT) + 255) / 256, 256, 0, stream>>>(P(70), BpD, LATENT, NUM_OUT);

  // encoders
  embed_k<<<((Nn * LATENT) + 255) / 256, 256, 0, stream>>>(node_feats, 9, P(6), P(7), nodes, Nn);
  embed_k<<<((Ne * LATENT) + 255) / 256, 256, 0, stream>>>(edge_feats, 3, P(8), P(9), edges, Ne);

  for (int t = 0; t < NSTEPS; ++t) {
    int base = 10 + t * 12;
    int gdim = (t == 0) ? NUM_OUT : LATENT;

    // --- edge update: concat(edges, nodes[snd], nodes[rcv], glob[egidx]) @ W
    gemm_fused<2><<<Ne / 32, 256, 0, stream>>>(
        edges, nullptr,   LATENT,
        nodes, senders,   LATENT,
        nodes, receivers, LATENT,
        glob,  edge_gidx, gdim,
        BpE[t], Hbuf, DinE[t]);
    ln_relu_k<<<Ne, 256, 0, stream>>>(Hbuf, P(base + 1), P(base + 2), P(base + 3), edges);

    // --- segment sums of updated edges
    fill_zero<<<((size_t)Nn * LATENT + 255) / 256, 256, 0, stream>>>(sacc, (size_t)Nn * LATENT);
    fill_zero<<<((size_t)Nn * LATENT + 255) / 256, 256, 0, stream>>>(racc, (size_t)Nn * LATENT);
    scatter_k<<<(int)(((size_t)Ne * LATENT) / 256), 256, 0, stream>>>(edges, senders, receivers, sacc, racc);

    // --- node update: concat(nodes, sent_agg, recv_agg, glob[ngidx]) @ W
    gemm_fused<2><<<Nn / 32, 256, 0, stream>>>(
        nodes, nullptr,   LATENT,
        sacc,  nullptr,   LATENT,
        racc,  nullptr,   LATENT,
        glob,  node_gidx, gdim,
        BpN[t], Hbuf, DinE[t]);
    ln_relu_k<<<Nn, 256, 0, stream>>>(Hbuf, P(base + 5), P(base + 6), P(base + 7), nodes);

    // --- global update: concat(node_agg, edge_agg, glob) @ W
    gsum_k<<<Gg, 256, 0, stream>>>(nodes, nagg, Nn / Gg);
    gsum_k<<<Gg, 256, 0, stream>>>(edges, eagg, Ne / Gg);
    gemm_fused<2><<<Gg / 32, 256, 0, stream>>>(
        nagg, nullptr, LATENT,
        eagg, nullptr, LATENT,
        glob, nullptr, gdim,
        nullptr, nullptr, 0,
        BpG[t], Hbuf, DinG[t]);
    ln_relu_k<<<Gg, 256, 0, stream>>>(Hbuf, P(base + 9), P(base + 10), P(base + 11), glob);
  }

  // decoder: glob[128,256] @ dec_W[256,128] + dec_b
  gemm_fused<1><<<Gg / 32, 256, 0, stream>>>(
      glob, nullptr, LATENT,
      nullptr, nullptr, 0,
      nullptr, nullptr, 0,
      nullptr, nullptr, 0,
      BpD, Hbuf, LATENT);
  bias_out_k<<<((Gg * NUM_OUT) + 255) / 256, 256, 0, stream>>>(Hbuf, P(71), (float*)d_out, Gg * NUM_OUT, NUM_OUT);
}
